// MixtureOfAdapter_58360015618070
// MI455X (gfx1250) — matrix operand order repaired and checked by reading, not compiled, not run on hardware
//
#include <hip/hip_runtime.h>
#include <hip/hip_bf16.h>
#include <math.h>

// Problem constants (fixed by the reference)
#define BB 8
#define LL 2048
#define HH 1024
#define FF 256
#define DD 3
#define NTOK (BB * LL)        // 16384
#define MBLK 32               // tokens per workgroup
#define NG1 (DD * FF)         // 768 stacked GEMM1 N / GEMM2 K
#define KC 64                 // K-chunk for GEMM1 staging

typedef __attribute__((ext_vector_type(16))) __bf16 v16bf;
typedef __attribute__((ext_vector_type(8)))  float  v8f;

union Frag16 {
    uint4 q[2];
    v16bf v;
};

static __device__ __forceinline__ unsigned short f2bf(float f) {
    union { float f; unsigned int u; } c; c.f = f;
    unsigned int u = c.u;
    u += 0x7FFFu + ((u >> 16) & 1u);   // round-to-nearest-even
    return (unsigned short)(u >> 16);
}

static __device__ __forceinline__ float bf2f(unsigned short s) {
    union { unsigned int u; float f; } c; c.u = ((unsigned int)s) << 16;
    return c.f;
}

static __device__ __forceinline__ v8f wmma_bf16(const Frag16& a, const Frag16& b, v8f c) {
    return __builtin_amdgcn_wmma_f32_16x16x32_bf16(
        /*neg_a=*/false, a.v, /*neg_b=*/false, b.v,
        /*c_mod=*/(short)0, c, /*reuse_a=*/false, /*reuse_b=*/false);
}

// ---------------------------------------------------------------------------
// Prep kernel 1: bf16 weight copies with LayerNorm-gamma folded into W1.
//   W1g[d,f,h] = bf16( ln_g[d,h] * W1[d,f,h] )   (B-matrix rows for GEMM1)
//   W2b[d,h,f] = bf16( W2[d,h,f] )               (B-matrix rows for GEMM2)
// ---------------------------------------------------------------------------
__global__ __launch_bounds__(256) void prep_weights(
    const float* __restrict__ W1, const float* __restrict__ ln_g,
    const float* __restrict__ W2,
    unsigned short* __restrict__ W1g, unsigned short* __restrict__ W2b)
{
    int i = blockIdx.x * 256 + threadIdx.x;       // 0 .. 786431
    int h  = i & (HH - 1);
    int df = i >> 10;                             // d*FF + f
    int d  = df >> 8;
    W1g[i] = f2bf(W1[i] * ln_g[d * HH + h]);
    W2b[i] = f2bf(W2[i]);
}

// ---------------------------------------------------------------------------
// Prep kernel 2: small reductions.
//  blocks [0,768):    c1[d,f]   = b1[d,f] + sum_h ln_b[d,h]*W1[d,f,h]
//  blocks [768,1536): w2gv[d,f] = sum_h gv[d,h]*W2[d,h,f]
//  block  1536:       bgate[d]  = gb[d] + sum_h b2[d,h]*gv[d,h]
// ---------------------------------------------------------------------------
__global__ __launch_bounds__(256) void prep_reduce(
    const float* __restrict__ W1, const float* __restrict__ ln_b,
    const float* __restrict__ b1, const float* __restrict__ W2,
    const float* __restrict__ gv, const float* __restrict__ b2,
    const float* __restrict__ gb,
    float* __restrict__ c1, float* __restrict__ w2gv, float* __restrict__ bgate)
{
    __shared__ float red[256];
    const int tid = threadIdx.x;
    const int blk = blockIdx.x;

    if (blk < 768) {
        int df = blk, d = df >> 8;
        float s = 0.f;
        for (int h = tid; h < HH; h += 256)
            s += ln_b[d * HH + h] * W1[(size_t)df * HH + h];
        red[tid] = s; __syncthreads();
        for (int st = 128; st > 0; st >>= 1) {
            if (tid < st) red[tid] += red[tid + st];
            __syncthreads();
        }
        if (tid == 0) c1[df] = red[0] + b1[df];
    } else if (blk < 1536) {
        int df = blk - 768, d = df >> 8, f = df & (FF - 1);
        float s = 0.f;
        for (int h = tid; h < HH; h += 256)
            s += gv[d * HH + h] * W2[((size_t)d * HH + h) * FF + f];
        red[tid] = s; __syncthreads();
        for (int st = 128; st > 0; st >>= 1) {
            if (tid < st) red[tid] += red[tid + st];
            __syncthreads();
        }
        if (tid == 0) w2gv[df] = red[0];
    } else {
        for (int d = 0; d < DD; ++d) {
            float s = 0.f;
            for (int h = tid; h < HH; h += 256)
                s += b2[d * HH + h] * gv[d * HH + h];
            red[tid] = s; __syncthreads();
            for (int st = 128; st > 0; st >>= 1) {
                if (tid < st) red[tid] += red[tid + st];
                __syncthreads();
            }
            if (tid == 0) bgate[d] = red[0] + gb[d];
            __syncthreads();
        }
    }
}

// ---------------------------------------------------------------------------
// Fused main kernel: stats -> GEMM1(stacked, bf16 WMMA) -> gate -> scale ->
// GEMM2(stacked over domains) -> epilogue (2x + gated adapter + gated b2).
// 256 threads (8 wave32), 32 tokens per workgroup.
// ---------------------------------------------------------------------------
__global__ __launch_bounds__(256) void moa_main(
    const float* __restrict__ x, const float* __restrict__ gu,
    const float* __restrict__ b2,
    const unsigned short* __restrict__ W1g, const unsigned short* __restrict__ W2b,
    const float* __restrict__ c1, const float* __restrict__ w2gv,
    const float* __restrict__ bgate, float* __restrict__ out)
{
    __shared__ unsigned short sh_mid[MBLK * NG1];   // 48 KB (aliased for gu in phase 0)
    __shared__ unsigned short sh_xn[MBLK * KC];     // 4 KB staged A chunk
    __shared__ float sh_mu[MBLK], sh_rsd[MBLK];
    __shared__ float sh_xg[DD][MBLK];
    __shared__ float sh_gacc[DD][MBLK];
    __shared__ float sh_gate[DD][MBLK];

    const int tid  = threadIdx.x;
    const int tok0 = blockIdx.x * MBLK;
    const int w    = tid >> 5;            // wave id 0..7
    const int ln   = tid & 15;            // lane-in-half (matrix col)
    const int half = (tid >> 4) & 1;      // half-wave (K split)

    // ---- phase 0a: preload gu into LDS scratch (aliasing sh_mid), zero gate acc
    float* gu_s = (float*)sh_mid;
    for (int i = tid; i < DD * HH; i += 256) gu_s[i] = gu[i];
    if (tid < DD * MBLK) (&sh_gacc[0][0])[tid] = 0.f;
    __syncthreads();

    // ---- phase 0b: per-token mean / unbiased std / x.gu_d  (8 threads / token)
    const int tl = tid >> 3, sb = tid & 7;
    {
        const float4* xr = (const float4*)(x + (size_t)(tok0 + tl) * HH);
        float s = 0.f, s2 = 0.f, g0 = 0.f, g1 = 0.f, g2 = 0.f;
        for (int j = 0; j < 32; ++j) {
            int q = sb + 8 * j;
            float4 v = xr[q];
            int h = q * 4;
            s  += v.x + v.y + v.z + v.w;
            s2 += v.x*v.x + v.y*v.y + v.z*v.z + v.w*v.w;
            g0 += v.x*gu_s[h] + v.y*gu_s[h+1] + v.z*gu_s[h+2] + v.w*gu_s[h+3];
            g1 += v.x*gu_s[HH+h] + v.y*gu_s[HH+h+1] + v.z*gu_s[HH+h+2] + v.w*gu_s[HH+h+3];
            g2 += v.x*gu_s[2*HH+h] + v.y*gu_s[2*HH+h+1] + v.z*gu_s[2*HH+h+2] + v.w*gu_s[2*HH+h+3];
        }
        #pragma unroll
        for (int m = 1; m < 8; m <<= 1) {
            s  += __shfl_xor(s,  m, 32);
            s2 += __shfl_xor(s2, m, 32);
            g0 += __shfl_xor(g0, m, 32);
            g1 += __shfl_xor(g1, m, 32);
            g2 += __shfl_xor(g2, m, 32);
        }
        if (sb == 0) {
            float mu  = s * (1.0f / (float)HH);
            float var = (s2 - (float)HH * mu * mu) * (1.0f / (float)(HH - 1));
            float sd  = sqrtf(fmaxf(var, 0.f));
            sh_mu[tl]  = mu;
            sh_rsd[tl] = 1.0f / (sd + 1e-6f);
            sh_xg[0][tl] = g0; sh_xg[1][tl] = g1; sh_xg[2][tl] = g2;
        }
    }
    __syncthreads();

    const v8f vzero = {0.f,0.f,0.f,0.f,0.f,0.f,0.f,0.f};

    // ---- GEMM1: mid[32,768] = xn[32,1024] @ W1g[768,1024]^T  (bf16 WMMA)
    v8f acc[2][6];
    #pragma unroll
    for (int mt = 0; mt < 2; ++mt)
        #pragma unroll
        for (int j = 0; j < 6; ++j) acc[mt][j] = vzero;

    const float muv = sh_mu[tl], rsv = sh_rsd[tl];
    for (int kb = 0; kb < HH; kb += KC) {
        __syncthreads();   // previous chunk's A-frag readers done
        {   // stage xn chunk (each thread: 8 contiguous K for its token)
            const float* xr = x + (size_t)(tok0 + tl) * HH + kb + sb * 8;
            unsigned short t[8];
            #pragma unroll
            for (int j = 0; j < 8; ++j) t[j] = f2bf((xr[j] - muv) * rsv);
            uint4 pk;
            pk.x = (unsigned)t[0] | ((unsigned)t[1] << 16);
            pk.y = (unsigned)t[2] | ((unsigned)t[3] << 16);
            pk.z = (unsigned)t[4] | ((unsigned)t[5] << 16);
            pk.w = (unsigned)t[6] | ((unsigned)t[7] << 16);
            *(uint4*)(sh_xn + tl * KC + sb * 8) = pk;
        }
        __syncthreads();

        #pragma unroll
        for (int ks = 0; ks < KC; ks += 32) {
            Frag16 a0, a1;   // A fragments: rows ln / ln+16, K-slice per half-wave
            const unsigned short* ap0 = sh_xn + ln * KC + ks + half * 8;
            a0.q[0] = *(const uint4*)ap0;
            a0.q[1] = *(const uint4*)(ap0 + 16);
            const unsigned short* ap1 = ap0 + 16 * KC;
            a1.q[0] = *(const uint4*)ap1;
            a1.q[1] = *(const uint4*)(ap1 + 16);

            #pragma unroll
            for (int j = 0; j < 6; ++j) {
                int nn = w * 96 + j * 16 + ln;   // stacked output column
                const unsigned short* bp =
                    W1g + (size_t)nn * HH + kb + ks + half * 8;
                Frag16 bfr;
                bfr.q[0] = *(const uint4*)bp;
                bfr.q[1] = *(const uint4*)(bp + 16);
                acc[0][j] = wmma_bf16(a0, bfr, acc[0][j]);
                acc[1][j] = wmma_bf16(a1, bfr, acc[1][j]);
            }
        }
    }

    // ---- GEMM1 epilogue: +c1, ReLU, store bf16 mid, gate partial (mid . w2gv)
    #pragma unroll
    for (int mt = 0; mt < 2; ++mt) {
        #pragma unroll
        for (int j = 0; j < 6; ++j) {
            int nnb = w * 96 + j * 16;
            int nn  = nnb + ln;
            int d   = nn >> 8;
            float c1v = c1[nn];
            float wgv = w2gv[nn];
            float gp[8];
            #pragma unroll
            for (int r = 0; r < 8; ++r) {
                float v = acc[mt][j][r] + c1v;
                v = fmaxf(v, 0.f);
                int m = mt * 16 + half * 8 + r;
                sh_mid[m * NG1 + nn] = f2bf(v);
                gp[r] = v * wgv;
            }
            #pragma unroll
            for (int msk = 1; msk < 16; msk <<= 1)
                #pragma unroll
                for (int r = 0; r < 8; ++r)
                    gp[r] += __shfl_xor(gp[r], msk, 32);
            if (ln == 0) {
                #pragma unroll
                for (int r = 0; r < 8; ++r)
                    atomicAdd(&sh_gacc[d][mt * 16 + half * 8 + r], gp[r]);
            }
        }
    }
    __syncthreads();

    // ---- gate = sigmoid(x.gu + mid.(W2^T gv) + b2.gv + gb)
    if (tid < DD * MBLK) {
        int d = tid >> 5, m = tid & 31;
        float t = sh_gacc[d][m] + sh_xg[d][m] + bgate[d];
        sh_gate[d][m] = 1.0f / (1.0f + __expf(-t));
    }
    __syncthreads();

    // ---- row-scale mid by its domain gate (folds gate into GEMM2's A)
    for (int idx = tid; idx < MBLK * NG1; idx += 256) {
        int m  = idx / NG1;
        int nn = idx - m * NG1;
        int d  = nn >> 8;
        sh_mid[idx] = f2bf(bf2f(sh_mid[idx]) * sh_gate[d][m]);
    }
    __syncthreads();

    // ---- GEMM2: out[32,1024] = sum over stacked K=768 of mid_scaled @ W2^T
    for (int j = 0; j < 8; ++j) {
        int hb = (w * 8 + j) * 16;
        int h  = hb + ln;
        v8f acc2[2];
        acc2[0] = vzero; acc2[1] = vzero;
        for (int dk = 0; dk < NG1; dk += 32) {
            Frag16 a0, a1;
            const unsigned short* ap0 = sh_mid + ln * NG1 + dk + half * 8;
            a0.q[0] = *(const uint4*)ap0;
            a0.q[1] = *(const uint4*)(ap0 + 16);
            const unsigned short* ap1 = ap0 + 16 * NG1;
            a1.q[0] = *(const uint4*)ap1;
            a1.q[1] = *(const uint4*)(ap1 + 16);

            int d = dk >> 8;
            const unsigned short* bp =
                W2b + ((size_t)(d * HH + h)) * FF + (dk & (FF - 1)) + half * 8;
            Frag16 bfr;
            bfr.q[0] = *(const uint4*)bp;
            bfr.q[1] = *(const uint4*)(bp + 16);
            acc2[0] = wmma_bf16(a0, bfr, acc2[0]);
            acc2[1] = wmma_bf16(a1, bfr, acc2[1]);
        }
        float b20 = b2[h], b21 = b2[HH + h], b22 = b2[2 * HH + h];
        #pragma unroll
        for (int mt = 0; mt < 2; ++mt) {
            #pragma unroll
            for (int r = 0; r < 8; ++r) {
                int m = mt * 16 + half * 8 + r;
                size_t off = (size_t)(tok0 + m) * HH + h;
                float bias = sh_gate[0][m] * b20 + sh_gate[1][m] * b21
                           + sh_gate[2][m] * b22;
                out[off] = 2.0f * x[off] + acc2[mt][r] + bias;
            }
        }
    }
}

// ---------------------------------------------------------------------------
extern "C" void kernel_launch(void* const* d_in, const int* in_sizes, int n_in,
                              void* d_out, int out_size, void* d_ws, size_t ws_size,
                              hipStream_t stream) {
    (void)in_sizes; (void)n_in; (void)out_size; (void)ws_size;
    const float* x    = (const float*)d_in[0];
    const float* ln_g = (const float*)d_in[1];
    const float* ln_b = (const float*)d_in[2];
    const float* W1   = (const float*)d_in[3];
    const float* b1   = (const float*)d_in[4];
    const float* W2   = (const float*)d_in[5];
    const float* b2   = (const float*)d_in[6];
    const float* gu   = (const float*)d_in[7];
    const float* gv   = (const float*)d_in[8];
    const float* gb   = (const float*)d_in[9];
    float* out = (float*)d_out;

    // workspace carve-up
    const size_t WELTS = (size_t)DD * FF * HH;     // 786432
    unsigned short* W1g = (unsigned short*)d_ws;
    unsigned short* W2b = W1g + WELTS;
    float* c1    = (float*)(W2b + WELTS);
    float* w2gv  = c1 + NG1;
    float* bgate = w2gv + NG1;

    prep_weights<<<(int)(WELTS / 256), 256, 0, stream>>>(W1, ln_g, W2, W1g, W2b);
    prep_reduce<<<2 * NG1 + 1, 256, 0, stream>>>(W1, ln_b, b1, W2, gv, b2, gb,
                                                 c1, w2gv, bgate);
    moa_main<<<NTOK / MBLK, 256, 0, stream>>>(x, gu, b2, W1g, W2b,
                                              c1, w2gv, bgate, out);
}
